// VQGAN_3779571221106
// MI455X (gfx1250) — matrix operand order repaired and checked by reading, not compiled, not run on hardware
//
#include <hip/hip_runtime.h>
#include <hip/hip_bf16.h>

// ---------------------------------------------------------------------------
// VQGAN forward on gfx1250: convs/deconvs + VQ scores as implicit GEMMs on
// v_wmma_f32_16x16x32_bf16 (wave32).  Weights are streamed into LDS with the
// Tensor Data Mover (TENSOR_LOAD_TO_LDS, double buffered, s_wait_tensorcnt);
// im2col A fragments are gathered from global with f32->bf16 convert.
// ---------------------------------------------------------------------------

typedef __attribute__((ext_vector_type(16))) __bf16 v16bf;
typedef __attribute__((ext_vector_type(8)))  float  v8f;
typedef __attribute__((ext_vector_type(4)))  unsigned int v4u;
typedef __attribute__((ext_vector_type(8)))  int v8i;
typedef __attribute__((ext_vector_type(4)))  int v4i;

// Swizzled B layout: tile (tileN, tileK) of 32x16 stored as 512 contiguous
// bf16: element index = lane*16 + i, lane = (n&15) | ((k>>3)&1)<<4,
// i = (k&7) | ((k>>4)&1)<<3.
// Matches the WMMA 16-bit B-matrix (32x16) VGPR layout: a lane's fragment is
// one contiguous 32-byte run.
__device__ __forceinline__ int bswz(int k, int n, int Kpad) {
  const int tileN = n >> 4, col = n & 15, tileK = k >> 5;
  const int hi = (k >> 3) & 1, i = (k & 7) | (((k >> 4) & 1) << 3);
  return ((tileN * (Kpad >> 5) + tileK) << 9) + ((col | (hi << 4)) << 4) + i;
}

// TDM: load a 2D tile (NT rows x 1024 bytes, row stride = rowStrideElems*4 B)
// from global into LDS at byte offset lds_byte.  D# per CDNA5 ISA ch.8:
//   group0: count=1, lds_addr, global_addr[56:0], type=2
//   group1: data_size=2 (4B), tensor_dim0/1 = 2^30 (no OOB clipping),
//           tile_dim0=256 elems, tile_dim1=rows, tensor_dim0_stride.
// This toolchain exposes the 6-arg builtin (g0, g1, g2, g3, g4, cpol).
__device__ __forceinline__ void tdm_load_b(const void* gsrc, unsigned lds_byte,
                                           int rows, int rowStrideElems) {
  const unsigned long long ga = (unsigned long long)(uintptr_t)gsrc;
  v4u g0;
  g0[0] = 1u;                                            // count = 1 valid D#
  g0[1] = lds_byte;                                      // lds_addr
  g0[2] = (unsigned)ga;                                  // global_addr[31:0]
  g0[3] = (unsigned)((ga >> 32) & 0x1FFFFFFu) | (2u << 30);  // ga[56:32]|type=2
  v8i g1;
  g1[0] = 0x20000;                 // data_size = 4 bytes
  g1[1] = 0;                       // tensor_dim0[15:0] (=0, dim0 = 2^30)
  g1[2] = 0x4000;                  // tensor_dim0[31:16] = 0x4000
  g1[3] = (256 << 16) | 0x4000;    // tile_dim0 = 256 | tensor_dim1[31:16]
  g1[4] = rows;                    // tile_dim1 (tile_dim2 = 0)
  g1[5] = rowStrideElems;          // tensor_dim0_stride[31:0]
  g1[6] = 0;
  g1[7] = 0;
  const v4i z4 = {0, 0, 0, 0};     // groups 2/3 unused (2D tensor)
  const v8i z8 = {0, 0, 0, 0, 0, 0, 0, 0};
  __builtin_amdgcn_tensor_load_to_lds(g0, g1, z4, z4, z8, 0);
}

// ---------------------------- weight packing -------------------------------

__global__ void pack_conv_w(const float* __restrict__ W, __bf16* __restrict__ Bsw,
                            int Cin, int Cout, int Kreal, int Kpad, int Npad) {
  const int t = blockIdx.x * blockDim.x + threadIdx.x;
  if (t >= Kpad * Npad) return;
  const int k = t / Npad, n = t - (t / Npad) * Npad;
  float v = 0.0f;
  if (k < Kreal && n < Cout) {
    const int ci = k >> 4, ky = (k >> 2) & 3, kx = k & 3;
    v = W[((n * Cin + ci) * 4 + ky) * 4 + kx];
  }
  Bsw[bswz(k, n, Kpad)] = (__bf16)v;
}

// Transposed conv k4/s2/p1: 4 taps per output; outputs split into 4 parity
// classes (py,px) = (oy&1, ox&1).  Tap j=(jy,jx) uses kernel (2jy+py,2jx+px)
// and input (y+jy-1+py, x+jx-1+px).
__global__ void pack_deconv_w(const float* __restrict__ W, __bf16* __restrict__ Bsw,
                              int Cin, int Cout, int Kpad, int Npad) {
  const int t = blockIdx.x * blockDim.x + threadIdx.x;
  const int per = Kpad * Npad;
  if (t >= 4 * per) return;
  const int cls = t / per, r = t - cls * per;
  const int k = r / Npad, n = r - (r / Npad) * Npad;
  const int py = cls >> 1, px = cls & 1;
  float v = 0.0f;
  const int ci = k >> 2, jy = (k >> 1) & 1, jx = k & 1;
  if (n < Cout)
    v = W[((n * Cin + ci) * 4 + (2 * jy + py)) * 4 + (2 * jx + px)];
  Bsw[(size_t)cls * ((Kpad >> 5) * (Npad >> 4) << 9) + bswz(k, n, Kpad)] = (__bf16)v;
}

__global__ void pack_cb(const float* __restrict__ cb, __bf16* __restrict__ Bsw) {
  const int t = blockIdx.x * blockDim.x + threadIdx.x;
  if (t >= 256 * 1024) return;
  const int k = t >> 10, n = t & 1023;
  const float v = (n < 1000) ? cb[n * 256 + k] : 0.0f;
  Bsw[bswz(k, n, 256)] = (__bf16)v;
}

__global__ void cb_norm(const float* __restrict__ cb, float* __restrict__ cnorm) {
  const int j = blockIdx.x * blockDim.x + threadIdx.x;
  if (j >= 1024) return;
  float s = 3.0e38f;  // padded columns never win argmin
  if (j < 1000) {
    s = 0.0f;
    for (int c = 0; c < 256; ++c) { const float v = cb[j * 256 + c]; s += v * v; }
  }
  cnorm[j] = s;
}

// ------------------------- implicit-GEMM conv (s2,k4,p1) -------------------

template <bool RELU, int NT>
__global__ __launch_bounds__(256)
void conv_igemm(const float* __restrict__ in, float* __restrict__ out,
                const __bf16* __restrict__ Bsw, const float* __restrict__ bias,
                int NB, int Cin, int H, int W, int Cout, int Npad,
                int Kreal, int Kpad) {
  __shared__ __bf16 ldsb[8 * 2 * NT * 512];    // 8 waves x 2 bufs x NT tiles
  const int OH = H >> 1, OW = W >> 1;
  const int lane = threadIdx.x & 31;
  const int wslot = threadIdx.x >> 5;
  const int wid  = blockIdx.x * (blockDim.x >> 5) + wslot;
  const int spat = OH * OW;
  const int mtiles = (NB * spat) >> 4;
  const int groups = Npad / (16 * NT);
  if (wid >= mtiles * groups) return;          // wave-uniform: EXEC stays full
  const int mtile = wid / groups;
  const int grp   = wid - mtile * groups;
  const int m0 = mtile << 4, n0 = grp * (16 * NT);

  const int hi8 = (lane >> 4) << 3;            // K-chunk offset per lane half
  const int m  = m0 + (lane & 15);             // A row owned by this lane
  const int nb = m / spat;
  const int rem = m - nb * spat;
  const int oy = rem / OW, ox = rem - (rem / OW) * OW;
  const int iy0 = 2 * oy - 1, ix0 = 2 * ox - 1;
  const float* inb = in + (size_t)nb * Cin * H * W;
  const int ktiles = Kpad >> 5;

  __bf16* myb = ldsb + wslot * (2 * NT * 512);
  const unsigned lds0 = (unsigned)(uintptr_t)(void*)myb;
  const __bf16* Bk = Bsw + ((size_t)(n0 >> 4) * ktiles << 9);

  // prologue: TDM-stream K-tile 0 into buffer 0
  tdm_load_b(Bk, lds0, NT, ktiles << 8);
  int buf = 0;

  v8f acc[NT] = {};
  for (int k0 = 0; k0 < Kpad; k0 += 32) {
    const bool last = (k0 + 32 >= Kpad);
    if (!last)  // TDM-stream next K-tile into the other buffer
      tdm_load_b(Bk + ((size_t)((k0 >> 5) + 1) << 9),
                 lds0 + (unsigned)((buf ^ 1) * NT * 1024), NT, ktiles << 8);
    v16bf a;
#pragma unroll
    for (int i = 0; i < 16; ++i) {
      // ISA 16-bit A 16x32 layout: elems 0..7 -> K+0..7 (+8 for hi lanes),
      // elems 8..15 -> K+16..23 (+8 for hi lanes)
      const int k = k0 + ((i & 8) << 1) + hi8 + (i & 7);
      float v = 0.0f;
      if (k < Kreal) {
        const int ci = k >> 4, ky = (k >> 2) & 3, kx = k & 3;
        const int iy = iy0 + ky, ix = ix0 + kx;
        if ((unsigned)iy < (unsigned)H && (unsigned)ix < (unsigned)W)
          v = inb[(ci * H + iy) * W + ix];
      }
      a[i] = (__bf16)v;
    }
    if (k0 + 64 < Kreal)                        // speculative A prefetch
      __builtin_prefetch(&inb[(((k0 + 64) >> 4) * H + iy0) * W + ix0], 0, 1);
    if (!last) __builtin_amdgcn_s_wait_tensorcnt(1);  // cur buffer ready
    else       __builtin_amdgcn_s_wait_tensorcnt(0);
#pragma unroll
    for (int t = 0; t < NT; ++t) {
      const v16bf b = *(const v16bf*)(myb + buf * (NT * 512) + t * 512 + (lane << 4));
      acc[t] = __builtin_amdgcn_wmma_f32_16x16x32_bf16(
          false, a, false, b, (short)0, acc[t], false, false);
    }
    buf ^= 1;
  }
#pragma unroll
  for (int t = 0; t < NT; ++t) {
    const int col = n0 + t * 16 + (lane & 15);
    if (col >= Cout) continue;
    const float bv = bias[col];
#pragma unroll
    for (int r = 0; r < 8; ++r) {               // C/D: VGPR r -> row r (+8 hi)
      const int mm = m0 + hi8 + r;
      const int nb2 = mm / spat;
      const int rem2 = mm - nb2 * spat;
      const int oy2 = rem2 / OW, ox2 = rem2 - (rem2 / OW) * OW;
      float v = acc[t][r] + bv;
      if (RELU) v = fmaxf(v, 0.0f);
      out[(((size_t)nb2 * Cout + col) * OH + oy2) * OW + ox2] = v;
    }
  }
}

// --------------------- implicit-GEMM transposed conv -----------------------

template <int ACT, int NT>  // ACT: 1 = relu, 2 = tanh
__global__ __launch_bounds__(256)
void deconv_igemm(const float* __restrict__ in, float* __restrict__ out,
                  const __bf16* __restrict__ Bsw, const float* __restrict__ bias,
                  int NB, int Cin, int S, int Cout, int Npad, int Kpad) {
  __shared__ __bf16 ldsb[8 * 2 * NT * 512];
  const int cls = blockIdx.y;                  // output parity class
  const int py = cls >> 1, px = cls & 1;
  const int OH = S << 1, OW = S << 1;
  const int lane = threadIdx.x & 31;
  const int wslot = threadIdx.x >> 5;
  const int wid  = blockIdx.x * (blockDim.x >> 5) + wslot;
  const int spat = S * S;
  const int mtiles = (NB * spat) >> 4;
  const int groups = Npad / (16 * NT);
  if (wid >= mtiles * groups) return;
  const int mtile = wid / groups;
  const int grp   = wid - mtile * groups;
  const int m0 = mtile << 4, n0 = grp * (16 * NT);

  const int hi8 = (lane >> 4) << 3;
  const int m = m0 + (lane & 15);
  const int nb = m / spat;
  const int rem = m - nb * spat;
  const int y = rem / S, x = rem - (rem / S) * S;
  const int iy0 = y - 1 + py, ix0 = x - 1 + px;
  const float* inb = in + (size_t)nb * Cin * spat;
  const int ktiles = Kpad >> 5;

  __bf16* myb = ldsb + wslot * (2 * NT * 512);
  const unsigned lds0 = (unsigned)(uintptr_t)(void*)myb;
  const __bf16* Bk = Bsw + ((size_t)cls * (ktiles * (Npad >> 4)) << 9)
                         + ((size_t)(n0 >> 4) * ktiles << 9);

  tdm_load_b(Bk, lds0, NT, ktiles << 8);
  int buf = 0;

  v8f acc[NT] = {};
  for (int k0 = 0; k0 < Kpad; k0 += 32) {
    const bool last = (k0 + 32 >= Kpad);
    if (!last)
      tdm_load_b(Bk + ((size_t)((k0 >> 5) + 1) << 9),
                 lds0 + (unsigned)((buf ^ 1) * NT * 1024), NT, ktiles << 8);
    v16bf a;
#pragma unroll
    for (int i = 0; i < 16; ++i) {
      const int k = k0 + ((i & 8) << 1) + hi8 + (i & 7);
      const int ci = k >> 2, jy = (k >> 1) & 1, jx = k & 1;
      const int iy = iy0 + jy, ix = ix0 + jx;
      float v = 0.0f;
      if ((unsigned)iy < (unsigned)S && (unsigned)ix < (unsigned)S)
        v = inb[(ci * S + iy) * S + ix];
      a[i] = (__bf16)v;
    }
    if (k0 + 64 < Kpad)
      __builtin_prefetch(&inb[(((k0 + 64) >> 2) * S + iy0) * S + ix0], 0, 1);
    if (!last) __builtin_amdgcn_s_wait_tensorcnt(1);
    else       __builtin_amdgcn_s_wait_tensorcnt(0);
#pragma unroll
    for (int t = 0; t < NT; ++t) {
      const v16bf b = *(const v16bf*)(myb + buf * (NT * 512) + t * 512 + (lane << 4));
      acc[t] = __builtin_amdgcn_wmma_f32_16x16x32_bf16(
          false, a, false, b, (short)0, acc[t], false, false);
    }
    buf ^= 1;
  }
#pragma unroll
  for (int t = 0; t < NT; ++t) {
    const int col = n0 + t * 16 + (lane & 15);
    if (col >= Cout) continue;
    const float bv = bias[col];
#pragma unroll
    for (int r = 0; r < 8; ++r) {
      const int mm = m0 + hi8 + r;
      const int nb2 = mm / spat;
      const int rem2 = mm - nb2 * spat;
      const int y2 = rem2 / S, x2 = rem2 - (rem2 / S) * S;
      float v = acc[t][r] + bv;
      if (ACT == 1) v = fmaxf(v, 0.0f);
      if (ACT == 2) v = tanhf(v);
      out[(((size_t)nb2 * Cout + col) * OH + (2 * y2 + py)) * OW + (2 * x2 + px)] = v;
    }
  }
}

// ------------------------------ VQ ----------------------------------------

// S[m, j] = sum_c z[m,c] * cb[j,c]; M=4096, N=1024(pad), K=256.
__global__ __launch_bounds__(256)
void vq_igemm(const float* __restrict__ zmap, float* __restrict__ Smat,
              const __bf16* __restrict__ Bsw) {
  const int lane = threadIdx.x & 31;
  const int wid = blockIdx.x * (blockDim.x >> 5) + (threadIdx.x >> 5);
  const int groups = 16;  // 1024 / (16*4)
  if (wid >= 256 * groups) return;
  const int mtile = wid / groups, grp = wid - (wid / groups) * groups;
  const int m0 = mtile << 4, n0 = grp * 64;
  const int hi8 = (lane >> 4) << 3;
  const int m = m0 + (lane & 15);
  // zmap NCHW [16,256,16,16]: row m=(nb,sp), elem c at nb*65536 + c*256 + sp
  const float* zb = zmap + ((size_t)(m >> 8) << 16) + (m & 255);
  const int ktiles = 8;
  v8f acc[4] = {};
  for (int k0 = 0; k0 < 256; k0 += 32) {
    v16bf a;
#pragma unroll
    for (int i = 0; i < 16; ++i) {
      const int k = k0 + ((i & 8) << 1) + hi8 + (i & 7);
      a[i] = (__bf16)zb[(size_t)k << 8];
    }
    const __bf16* bp = Bsw + ((size_t)((n0 >> 4) * ktiles + (k0 >> 5)) << 9) + (lane << 4);
#pragma unroll
    for (int t = 0; t < 4; ++t) {
      const v16bf b = *(const v16bf*)(bp + ((size_t)t * ktiles << 9));
      acc[t] = __builtin_amdgcn_wmma_f32_16x16x32_bf16(
          false, a, false, b, (short)0, acc[t], false, false);
    }
  }
#pragma unroll
  for (int t = 0; t < 4; ++t) {
    const int col = n0 + t * 16 + (lane & 15);
#pragma unroll
    for (int r = 0; r < 8; ++r)
      Smat[((size_t)(m0 + hi8 + r) << 10) + col] = acc[t][r];
  }
}

__global__ void vq_argmin(const float* __restrict__ Smat,
                          const float* __restrict__ cnorm, int* __restrict__ idx) {
  const int m = blockIdx.x * blockDim.x + threadIdx.x;
  if (m >= 4096) return;
  const float* row = Smat + ((size_t)m << 10);
  float best = 3.4e38f; int bi = 0;
  for (int j = 0; j < 1000; ++j) {
    const float d = cnorm[j] - 2.0f * row[j];   // |z|^2 is row-constant
    if (d < best) { best = d; bi = j; }
  }
  idx[m] = bi;
}

__global__ void vq_finalize(const float* __restrict__ zmap, const float* __restrict__ cb,
                            const int* __restrict__ idx, float* __restrict__ z_out,
                            float* __restrict__ e_out, float* __restrict__ zq) {
  const int t = blockIdx.x * blockDim.x + threadIdx.x;
  if (t >= 4096 * 256) return;
  const int m = t >> 8, c = t & 255;
  const int nb = m >> 8, sp = m & 255;
  const float zv = zmap[((size_t)nb << 16) + (c << 8) + sp];
  z_out[t] = zv;
  const float ev = cb[idx[m] * 256 + c];
  e_out[t] = ev;
  zq[((size_t)nb << 16) + (c << 8) + sp] = ev;  // NCHW for decoder
}

// ------------------------------ host side ----------------------------------

extern "C" void kernel_launch(void* const* d_in, const int* in_sizes, int n_in,
                              void* d_out, int out_size, void* d_ws, size_t ws_size,
                              hipStream_t stream) {
  (void)in_sizes; (void)n_in; (void)out_size; (void)ws_size;
  const float* x   = (const float*)d_in[0];
  const float* cb  = (const float*)d_in[1];
  const float* ew1 = (const float*)d_in[2];  const float* eb1 = (const float*)d_in[3];
  const float* ew2 = (const float*)d_in[4];  const float* eb2 = (const float*)d_in[5];
  const float* ew3 = (const float*)d_in[6];  const float* eb3 = (const float*)d_in[7];
  const float* ew4 = (const float*)d_in[8];  const float* eb4 = (const float*)d_in[9];
  const float* dw1 = (const float*)d_in[10]; const float* db1 = (const float*)d_in[11];
  const float* dw2 = (const float*)d_in[12]; const float* db2 = (const float*)d_in[13];
  const float* dw3 = (const float*)d_in[14]; const float* db3 = (const float*)d_in[15];
  const float* dw4 = (const float*)d_in[16]; const float* db4 = (const float*)d_in[17];

  float* xhat  = (float*)d_out;                      // [16,3,256,256]
  float* z_out = xhat + (size_t)16 * 3 * 256 * 256;  // [4096,256]
  float* e_out = z_out + (size_t)4096 * 256;         // [4096,256]

  char* ws = (char*)d_ws;
  size_t off = 0;
  auto alloc = [&](size_t bytes) -> char* {
    char* p = ws + off;
    off = (off + bytes + 255) & ~(size_t)255;
    return p;
  };

  float* bufA = (float*)alloc((size_t)16 * 64 * 128 * 128 * 4);  // a1 / Smat / g3
  float* bufB = (float*)alloc((size_t)16 * 128 * 64 * 64 * 4);   // a2 / g2
  float* bufC = (float*)alloc((size_t)16 * 256 * 32 * 32 * 4);   // a3 / g1
  float* zmap = (float*)alloc((size_t)4096 * 256 * 4);
  float* zq   = (float*)alloc((size_t)4096 * 256 * 4);
  int*   idx  = (int*)alloc(4096 * 4);
  float* cnrm = (float*)alloc(1024 * 4);
  __bf16* wc1 = (__bf16*)alloc((size_t)64 * 64 * 2);
  __bf16* wc2 = (__bf16*)alloc((size_t)1024 * 128 * 2);
  __bf16* wc3 = (__bf16*)alloc((size_t)2048 * 256 * 2);
  __bf16* wc4 = (__bf16*)alloc((size_t)4096 * 256 * 2);
  __bf16* wd1 = (__bf16*)alloc((size_t)4 * 1024 * 256 * 2);
  __bf16* wd2 = (__bf16*)alloc((size_t)4 * 1024 * 128 * 2);
  __bf16* wd3 = (__bf16*)alloc((size_t)4 * 512 * 64 * 2);
  __bf16* wd4 = (__bf16*)alloc((size_t)4 * 256 * 16 * 2);
  __bf16* wcb = (__bf16*)alloc((size_t)256 * 1024 * 2);
  float* Smat = bufA;  // alias: a1 is dead between conv2 and deconv3

  auto cdiv = [](long a, long b) { return (int)((a + b - 1) / b); };

  // ---- pack weights / codebook (tiny) ----
  pack_conv_w<<<cdiv(64 * 64, 256), 256, 0, stream>>>(ew1, wc1, 3, 64, 48, 64, 64);
  pack_conv_w<<<cdiv(1024 * 128, 256), 256, 0, stream>>>(ew2, wc2, 64, 128, 1024, 1024, 128);
  pack_conv_w<<<cdiv(2048 * 256, 256), 256, 0, stream>>>(ew3, wc3, 128, 256, 2048, 2048, 256);
  pack_conv_w<<<cdiv(4096 * 256, 256), 256, 0, stream>>>(ew4, wc4, 256, 256, 4096, 4096, 256);
  pack_deconv_w<<<cdiv(4L * 1024 * 256, 256), 256, 0, stream>>>(dw1, wd1, 256, 256, 1024, 256);
  pack_deconv_w<<<cdiv(4L * 1024 * 128, 256), 256, 0, stream>>>(dw2, wd2, 256, 128, 1024, 128);
  pack_deconv_w<<<cdiv(4L * 512 * 64, 256), 256, 0, stream>>>(dw3, wd3, 128, 64, 512, 64);
  pack_deconv_w<<<cdiv(4L * 256 * 16, 256), 256, 0, stream>>>(dw4, wd4, 64, 3, 256, 16);
  pack_cb<<<cdiv(256 * 1024, 256), 256, 0, stream>>>(cb, wcb);
  cb_norm<<<4, 256, 0, stream>>>(cb, cnrm);

  // ---- encoder ----
  conv_igemm<true, 4><<<cdiv(16384L * 1, 8), 256, 0, stream>>>(
      x, bufA, wc1, eb1, 16, 3, 256, 256, 64, 64, 48, 64);
  conv_igemm<true, 4><<<cdiv(4096L * 2, 8), 256, 0, stream>>>(
      bufA, bufB, wc2, eb2, 16, 64, 128, 128, 128, 128, 1024, 1024);
  conv_igemm<true, 4><<<cdiv(1024L * 4, 8), 256, 0, stream>>>(
      bufB, bufC, wc3, eb3, 16, 128, 64, 64, 256, 256, 2048, 2048);
  conv_igemm<false, 4><<<cdiv(256L * 4, 8), 256, 0, stream>>>(
      bufC, zmap, wc4, eb4, 16, 256, 32, 32, 256, 256, 4096, 4096);

  // ---- vector quantization ----
  vq_igemm<<<cdiv(256L * 16, 8), 256, 0, stream>>>(zmap, Smat, wcb);
  vq_argmin<<<16, 256, 0, stream>>>(Smat, cnrm, idx);
  vq_finalize<<<cdiv(4096L * 256, 256), 256, 0, stream>>>(zmap, cb, idx, z_out, e_out, zq);

  // ---- decoder (4 parity classes in grid.y) ----
  deconv_igemm<1, 4><<<dim3(cdiv(256L * 4, 8), 4), 256, 0, stream>>>(
      zq, bufC, wd1, db1, 16, 256, 16, 256, 256, 1024);
  deconv_igemm<1, 4><<<dim3(cdiv(1024L * 2, 8), 4), 256, 0, stream>>>(
      bufC, bufB, wd2, db2, 16, 256, 32, 128, 128, 1024);
  deconv_igemm<1, 4><<<dim3(cdiv(4096L * 1, 8), 4), 256, 0, stream>>>(
      bufB, bufA, wd3, db3, 16, 128, 64, 64, 64, 512);
  deconv_igemm<2, 1><<<dim3(cdiv(16384L * 1, 8), 4), 256, 0, stream>>>(
      bufA, xhat, wd4, db4, 16, 64, 128, 3, 16, 256);
}
